// LstmDecoder_71347996721772
// MI455X (gfx1250) — matrix operand order repaired and checked by reading, not compiled, not run on hardware
//
#include <hip/hip_runtime.h>

typedef _Float16 v16h __attribute__((ext_vector_type(16)));
typedef _Float16 v8h  __attribute__((ext_vector_type(8)));
typedef float    v8f  __attribute__((ext_vector_type(8)));

#define TDIM 256
#define FDIM 256
#define DDIM 128
#define EDIM 260      // F + 4
#define EPAD 288      // padded to 9 K-tiles of 32
#define NG   512      // 4*D gate columns
#define KT_X 9        // EPAD/32
#define KT_H 4        // D/32
#define OUTC 260      // output feature columns

__device__ __forceinline__ float fast_sigmoid(float x) {
    return 1.0f / (1.0f + __expf(-x));
}
__device__ __forceinline__ float fast_tanh(float x) {
    float e = __expf(2.0f * x);
    return (e - 1.0f) / (e + 1.0f);
}

// ---------------------------------------------------------------------------
// Pack W_ih (512 x 260 f32) into f16 B-fragments, K padded to 288.
// Fragment (kt,nt) covers k in [kt*32,kt*32+32), n in [nt*16,nt*16+16).
// Element order inside fragment: lane*16 + j; lane: n = nt*16 + (lane&15),
// khalf = lane>>4; j: k = kt*32 + khalf*16 + j.
// ---------------------------------------------------------------------------
__global__ void pack_wih(const float* __restrict__ W, _Float16* __restrict__ out) {
    int idx = blockIdx.x * blockDim.x + threadIdx.x;
    if (idx >= KT_X * 32 * 512) return;
    int frag   = idx >> 9;
    int within = idx & 511;
    int lane = within >> 4;
    int j    = within & 15;
    int kt = frag >> 5;
    int nt = frag & 31;
    int n = nt * 16 + (lane & 15);
    int k = kt * 32 + (lane >> 4) * 16 + j;
    float v = (k < EDIM) ? W[n * EDIM + k] : 0.0f;
    out[idx] = (_Float16)v;
}

__global__ void pack_whh(const float* __restrict__ W, _Float16* __restrict__ out) {
    int idx = blockIdx.x * blockDim.x + threadIdx.x;
    if (idx >= KT_H * 32 * 512) return;
    int frag   = idx >> 9;
    int within = idx & 511;
    int lane = within >> 4;
    int j    = within & 15;
    int kt = frag >> 5;
    int nt = frag & 31;
    int n = nt * 16 + (lane & 15);
    int k = kt * 32 + (lane >> 4) * 16 + j;
    out[idx] = (_Float16)W[n * DDIM + k];
}

__global__ void pack_bias(const float* __restrict__ b_ih, const float* __restrict__ b_hh,
                          float* __restrict__ bias) {
    int n = blockIdx.x * blockDim.x + threadIdx.x;
    if (n < NG) bias[n] = b_ih[n] + b_hh[n];
}

// Static output columns 256..259: shapes, gap_means, slopes (unmasked broadcast).
__global__ void write_static(const float* __restrict__ shapes, const float* __restrict__ gaps,
                             const float* __restrict__ slopes, float* __restrict__ out) {
    int idx = blockIdx.x * blockDim.x + threadIdx.x;   // over B*T
    if (idx >= 1024 * TDIM) return;
    int b = idx >> 8;                                  // / T
    float* o = out + (size_t)idx * OUTC;
    o[256] = shapes[idx * 2 + 0];
    o[257] = shapes[idx * 2 + 1];
    o[258] = gaps[b];
    o[259] = slopes[b];
}

// ---------------------------------------------------------------------------
// Fused bidirectional LSTM recurrence. blockIdx.x -> batch tile of 16 rows,
// blockIdx.y -> direction. 8 waves; wave w owns hidden columns [16w,16w+16):
// its four 16x16 accumulator tiles are exactly the i/f/g/o gate slices for
// those columns, so the cell update is wave-local.
//
// K-tile loops are kept ROLLED (#pragma unroll 1) so the 52 weight-fragment
// loads per step cannot be hoisted out of the t-loop by LICM (that caused
// >256-VGPR allocation + scratch spills). Weights re-stream from L2 (~0.43 MB
// packed, permanently L2-resident).
//
// x_t fetch is software-pipelined: step t issues next-step global loads
// before the GEMM, commits them to LDS after the mid-barrier.
// ---------------------------------------------------------------------------
__global__ __launch_bounds__(256)
void lstm_recur(const float* __restrict__ feat, const float* __restrict__ shapes,
                const float* __restrict__ gaps, const float* __restrict__ slopes,
                const int* __restrict__ n_ims,
                const _Float16* __restrict__ wih, const _Float16* __restrict__ whh,
                const float* __restrict__ bias,
                const float* __restrict__ init_h, const float* __restrict__ init_c,
                const float* __restrict__ init_h_rev, const float* __restrict__ init_c_rev,
                float* __restrict__ out)
{
    const int dir  = blockIdx.y;
    const int b0   = blockIdx.x * 16;
    const int tid  = threadIdx.x;
    const int wave = tid >> 5;
    const int lane = tid & 31;

    __shared__ _Float16 xbuf[16][EPAD];   // 9 KB
    __shared__ _Float16 hbuf[16][DDIM];   // 4 KB

    const float* ih = dir ? init_h_rev : init_h;
    const float* ic = dir ? init_c_rev : init_c;

    // h state to LDS (f16), static xbuf tail (cols 256..287): gap/slope/pad.
    for (int i = tid; i < 16 * DDIM; i += 256)
        hbuf[i >> 7][i & 127] = (_Float16)ih[i & 127];
    for (int i = tid; i < 16 * 32; i += 256) {
        int m = i >> 5;
        int e = FDIM + (i & 31);
        float v = 0.0f;
        if (e == FDIM + 2) v = gaps[b0 + m];
        else if (e == FDIM + 3) v = slopes[b0 + m];
        xbuf[m][e] = (_Float16)v;
    }

    const int nsub = lane & 15;           // column within 16-wide tile / A row M
    const int mgrp = lane >> 4;           // accumulator row-group (M offset 0/8)
    const int col  = wave * 16 + nsub;    // hidden column d in [0,128)

    // c/h carry state, register layout mirrors the WMMA f32 C/D layout:
    // VGPR r, this lane -> element (M = r + 8*mgrp, N = col).
    float creg[8], hreg[8];
    int   lens[8];
    #pragma unroll
    for (int r = 0; r < 8; ++r) {
        creg[r] = ic[col];
        hreg[r] = ih[col];
        lens[r] = n_ims[b0 + r + mgrp * 8];
    }

    float biasv[4];
    #pragma unroll
    for (int g = 0; g < 4; ++g) biasv[g] = bias[(g * 8 + wave) * 16 + nsub];

    // ---- software-pipelined x_t staging ----
    // thread tid owns feature column tid for all 16 batch rows.
    const size_t frow = (size_t)TDIM * FDIM;                // 65536
    const int fe = dir ? (FDIM - 1 - tid) : tid;
    const float* fbase = feat + (size_t)b0 * frow + fe;
    const int sj  = tid & 1;                                // dest shape column
    const int sse = dir ? 1 - sj : sj;                      // source shape column
    const float* sbase = shapes + (size_t)(b0 + (tid >> 1)) * TDIM * 2 + sse;

    float xstage[16];
    float sstage = 0.0f;

    auto stage_load = [&](int t) {
        #pragma unroll
        for (int m = 0; m < 16; ++m)
            xstage[m] = fbase[(size_t)m * frow + (size_t)t * FDIM];
        if (tid < 32) sstage = sbase[t * 2];
    };
    auto stage_commit = [&]() {
        #pragma unroll
        for (int m = 0; m < 16; ++m)
            xbuf[m][tid] = (_Float16)xstage[m];
        if (tid < 32) xbuf[tid >> 1][FDIM + sj] = (_Float16)sstage;
    };

    // output base: per r,t -> obase[r*T*OUTC + t*OUTC]
    const int ocol = dir ? (2 * DDIM - 1 - col) : col;
    float* obase = out + (size_t)(b0 + mgrp * 8) * TDIM * OUTC + ocol;

    stage_load(0);
    stage_commit();
    __syncthreads();

    for (int t = 0; t < TDIM; ++t) {
        // issue next step's global loads; consumed after the mid-barrier
        const int tn = (t + 1 < TDIM) ? t + 1 : t;
        stage_load(tn);

        v8f acc[4];
        #pragma unroll
        for (int g = 0; g < 4; ++g)
            #pragma unroll
            for (int r = 0; r < 8; ++r) acc[g][r] = biasv[g];

        // gates += x_t @ W_ih^T   (9 K-tiles, rolled)
        #pragma unroll 1
        for (int kt = 0; kt < KT_X; ++kt) {
            union { v16h v; v8h h[2]; } a;
            a.h[0] = *(const v8h*)&xbuf[nsub][kt * 32 + mgrp * 16];
            a.h[1] = *(const v8h*)&xbuf[nsub][kt * 32 + mgrp * 16 + 8];
            #pragma unroll
            for (int g = 0; g < 4; ++g) {
                int frag = kt * 32 + (g * 8 + wave);
                v16h bm = *(const v16h*)(wih + ((size_t)frag << 9) + (lane << 4));
                acc[g] = __builtin_amdgcn_wmma_f32_16x16x32_f16(
                    false, a.v, false, bm, (short)0, acc[g], false, false);
            }
        }
        // gates += h @ W_hh^T     (4 K-tiles, rolled)
        #pragma unroll 1
        for (int kt = 0; kt < KT_H; ++kt) {
            union { v16h v; v8h h[2]; } a;
            a.h[0] = *(const v8h*)&hbuf[nsub][kt * 32 + mgrp * 16];
            a.h[1] = *(const v8h*)&hbuf[nsub][kt * 32 + mgrp * 16 + 8];
            #pragma unroll
            for (int g = 0; g < 4; ++g) {
                int frag = kt * 32 + (g * 8 + wave);
                v16h bm = *(const v16h*)(whh + ((size_t)frag << 9) + (lane << 4));
                acc[g] = __builtin_amdgcn_wmma_f32_16x16x32_f16(
                    false, a.v, false, bm, (short)0, acc[g], false, false);
            }
        }
        __syncthreads();   // all xbuf/hbuf reads complete

        // LSTM cell update + masked output write (wave-local).
        #pragma unroll
        for (int r = 0; r < 8; ++r) {
            int m = r + mgrp * 8;
            float iv = fast_sigmoid(acc[0][r]);
            float fv = fast_sigmoid(acc[1][r]);
            float gv = fast_tanh(acc[2][r]);
            float ov = fast_sigmoid(acc[3][r]);
            float nc = fv * creg[r] + iv * gv;
            float nh = ov * fast_tanh(nc);
            bool active = t < lens[r];
            float outv = active ? nh : 0.0f;
            if (active) { creg[r] = nc; hreg[r] = nh; }
            hbuf[m][col] = (_Float16)hreg[r];
            obase[(size_t)r * TDIM * OUTC + (size_t)t * OUTC] = outv;
        }
        // commit staged x_{t+1} into xbuf
        stage_commit();
        __syncthreads();   // guards hbuf/xbuf writes vs next-step reads
    }
}

extern "C" void kernel_launch(void* const* d_in, const int* in_sizes, int n_in,
                              void* d_out, int out_size, void* d_ws, size_t ws_size,
                              hipStream_t stream) {
    (void)in_sizes; (void)n_in; (void)out_size; (void)ws_size;
    const float* feat       = (const float*)d_in[0];
    const float* shapes     = (const float*)d_in[1];
    const float* gaps       = (const float*)d_in[2];
    const float* slopes     = (const float*)d_in[3];
    const int*   n_ims      = (const int*)d_in[4];
    const float* W_ih       = (const float*)d_in[5];
    const float* W_hh       = (const float*)d_in[6];
    const float* b_ih       = (const float*)d_in[7];
    const float* b_hh       = (const float*)d_in[8];
    const float* init_h     = (const float*)d_in[9];
    const float* init_c     = (const float*)d_in[10];
    const float* init_h_rev = (const float*)d_in[11];
    const float* init_c_rev = (const float*)d_in[12];
    float* out = (float*)d_out;

    char* ws = (char*)d_ws;
    const size_t wih_bytes = (size_t)KT_X * 32 * 512 * sizeof(_Float16); // 294912
    const size_t whh_bytes = (size_t)KT_H * 32 * 512 * sizeof(_Float16); // 131072
    _Float16* wih_pk = (_Float16*)ws;
    _Float16* whh_pk = (_Float16*)(ws + wih_bytes);
    float*    biasd  = (float*)(ws + wih_bytes + whh_bytes);

    pack_wih<<<(KT_X * 32 * 512 + 255) / 256, 256, 0, stream>>>(W_ih, wih_pk);
    pack_whh<<<(KT_H * 32 * 512 + 255) / 256, 256, 0, stream>>>(W_hh, whh_pk);
    pack_bias<<<(NG + 255) / 256, 256, 0, stream>>>(b_ih, b_hh, biasd);
    write_static<<<(1024 * TDIM + 255) / 256, 256, 0, stream>>>(shapes, gaps, slopes, out);

    dim3 grid(1024 / 16, 2);
    lstm_recur<<<grid, 256, 0, stream>>>(feat, shapes, gaps, slopes, n_ims,
                                         wih_pk, whh_pk, biasd,
                                         init_h, init_c, init_h_rev, init_c_rev, out);
}
